// EdgeConv_10024453668967
// MI455X (gfx1250) — matrix operand order repaired and checked by reading, not compiled, not run on hardware
//
#include <hip/hip_runtime.h>

// EdgeConv (DGCNN) for MI455X / gfx1250, wave32 + WMMA f16->f32 + TDM staging.
// B=8, C=64, N=4096, O=128, K=20.

#define B_ 8
#define C_ 64
#define N_ 4096
#define O_ 128
#define K_ 20

typedef __attribute__((ext_vector_type(16))) _Float16 v16h;
typedef __attribute__((ext_vector_type(8)))  _Float16 v8h;
typedef __attribute__((ext_vector_type(8)))  float    v8f;

typedef __attribute__((ext_vector_type(4))) unsigned int u32x4;
typedef __attribute__((ext_vector_type(8))) int          i32x8;
typedef __attribute__((ext_vector_type(4))) int          i32x4;

union Frag16 { v16h v; v8h h[2]; };

// ---------------------------------------------------------------------------
// Prep: transpose x (B,C,N) -> xt32 (B,N,C) f32 + xth f16, and row norms xx.
// ---------------------------------------------------------------------------
__global__ void prep_x_kernel(const float* __restrict__ x,
                              float* __restrict__ xt32,
                              _Float16* __restrict__ xth,
                              float* __restrict__ xx) {
  int t = blockIdx.x * blockDim.x + threadIdx.x;   // 0 .. B*N-1
  int b = t / N_;
  int n = t - b * N_;
  const float* xp = x + (size_t)b * C_ * N_ + n;
  float*     op = xt32 + (size_t)t * C_;
  _Float16*  hp = xth  + (size_t)t * C_;
  float s = 0.f;
  #pragma unroll 8
  for (int c = 0; c < C_; ++c) {
    float v = xp[(size_t)c * N_];
    op[c] = v;
    hp[c] = (_Float16)v;
    s += v * v;
  }
  xx[t] = s;
}

__global__ void prep_w_kernel(const float* __restrict__ W,
                              _Float16* __restrict__ wh) {
  int t = blockIdx.x * blockDim.x + threadIdx.x;   // 0 .. O*2C-1
  wh[t] = (_Float16)W[t];
}

// ---------------------------------------------------------------------------
// KNN: one wave per 32 rows (two 16x16 A-tile sets sharing B fragments).
// WMMA 16x16 Gram tiles (K=64 -> 2 x 16x16x32_f16); selection key
// s = 2*G[i][j] - xx[j] (== pd[i][j] + xx[i], same ordering per row).
// All 32 lanes select: lane hi-half owns the second row tile.
// Top-20 kept sorted in registers via fully-unrolled bubble insertion.
// ---------------------------------------------------------------------------
__global__ __launch_bounds__(32) void knn_kernel(const _Float16* __restrict__ xth,
                                                 const float* __restrict__ xx,
                                                 int* __restrict__ knn) {
  __shared__ float sTile[2 * 16 * 16];

  const int b    = blockIdx.y;
  const int m0   = blockIdx.x * 32;
  const int lane = threadIdx.x;
  const int hi   = lane >> 4;     // half-wave index (0/1)
  const int lo   = lane & 15;

  const _Float16* xb = xth + (size_t)b * N_ * C_;

  // A fragments: two row tiles (m0, m0+16), K = 0..63 each (2 x 16x32 frags).
  Frag16 a[2][2];
  #pragma unroll
  for (int tile = 0; tile < 2; ++tile) {
    const _Float16* rp = xb + (size_t)(m0 + tile * 16 + lo) * C_;
    a[tile][0].h[0] = *(const v8h*)(rp +      hi * 8);
    a[tile][0].h[1] = *(const v8h*)(rp + 16 + hi * 8);
    a[tile][1].h[0] = *(const v8h*)(rp + 32 + hi * 8);
    a[tile][1].h[1] = *(const v8h*)(rp + 48 + hi * 8);
  }

  float tv[K_]; int ti[K_];
  #pragma unroll
  for (int t = 0; t < K_; ++t) { tv[t] = -3.0e38f; ti[t] = 0; }

  for (int j0 = 0; j0 < N_; j0 += 16) {
    // B fragments: columns j0..j0+15 (rows of Xt), table-order K (8-half runs).
    const _Float16* cp = xb + (size_t)(j0 + lo) * C_;
    Frag16 b0, b1;
    b0.h[0] = *(const v8h*)(cp +      hi * 8);
    b0.h[1] = *(const v8h*)(cp + 16 + hi * 8);
    b1.h[0] = *(const v8h*)(cp + 32 + hi * 8);
    b1.h[1] = *(const v8h*)(cp + 48 + hi * 8);

    float xxj = xx[b * N_ + j0 + lo];          // column norm (per lane)

    #pragma unroll
    for (int tile = 0; tile < 2; ++tile) {
      v8f c = {};
      c = __builtin_amdgcn_wmma_f32_16x16x32_f16(false, a[tile][0].v, false, b0.v,
                                                 (short)0, c, false, false);
      c = __builtin_amdgcn_wmma_f32_16x16x32_f16(false, a[tile][1].v, false, b1.v,
                                                 (short)0, c, false, false);
      #pragma unroll
      for (int r = 0; r < 8; ++r)
        sTile[tile * 256 + (r + 8 * hi) * 16 + lo] = 2.0f * c[r] - xxj;
    }
    __syncthreads();

    // Lane l owns global row m0 + hi*16 + lo -> reads tile `hi`, row `lo`.
    #pragma unroll
    for (int jj = 0; jj < 16; ++jj) {
      float v = sTile[hi * 256 + lo * 16 + jj];
      int   j = j0 + jj;
      if (v > tv[K_ - 1]) {
        #pragma unroll
        for (int t = 0; t < K_; ++t) {         // sorted insert, strict > keeps
          bool  g  = v > tv[t];                // lowest index on ties (jax)
          float ov = tv[t]; int oj = ti[t];
          tv[t] = g ? v : ov;  ti[t] = g ? j : oj;
          v     = g ? ov : v;  j     = g ? oj : j;
        }
      }
    }
    __syncthreads();
  }

  int* op = knn + ((size_t)b * N_ + m0 + hi * 16 + lo) * K_;
  #pragma unroll
  for (int t = 0; t < K_; ++t) op[t] = ti[t];
}

// ---------------------------------------------------------------------------
// Edge MLP: one wave per 16 points. W (128x128 f16, 32KB) is staged into LDS
// once per workgroup via the Tensor Data Mover (flat 1x16384 2-byte tile),
// synchronized with s_wait_tensorcnt. Edge features (concat[nbr-ctr, ctr])
// staged in LDS as f16 in two 10-slot halves (40KB). Per neighbor slot:
// 8 output tiles x 4 WMMA (K=128), bias + leaky + running max in D layout.
// ---------------------------------------------------------------------------
__global__ __launch_bounds__(32) void edge_kernel(const float* __restrict__ xt32,
                                                  const _Float16* __restrict__ wh,
                                                  const float* __restrict__ bias,
                                                  const int* __restrict__ knn,
                                                  float* __restrict__ out) {
  __shared__ _Float16 wLds[O_ * 2 * C_];      // 32 KB
  __shared__ _Float16 eLds[10][16][2 * C_];   // 40 KB
  __shared__ int nIdx[16 * K_];

  const int b    = blockIdx.y;
  const int m0   = blockIdx.x * 16;
  const int lane = threadIdx.x;
  const int hi   = lane >> 4;
  const int lo   = lane & 15;

#if defined(__has_builtin) && __has_builtin(__builtin_amdgcn_tensor_load_to_lds)
  {
    // D# group 0: count=1, lds_addr, global_addr, type=2 ("image").
    unsigned long long ga = (unsigned long long)(size_t)wh;
    u32x4 g0;
    g0[0] = 1u;                                            // count=1
    g0[1] = (unsigned int)(size_t)(&wLds[0]);              // LDS byte offset
    g0[2] = (unsigned int)(ga & 0xFFFFFFFFu);              // global_addr[31:0]
    g0[3] = (unsigned int)((ga >> 32) & 0x01FFFFFFu)       // global_addr[56:32]
            | (2u << 30);                                  // type=2
    // D# group 1: data_size=2B; tensor 16384x1; tile 16384x1; strides 16384.
    i32x8 g1;
    g1[0] = 0x00010000;             // workgroup_mask=0, data_size=1 (2 bytes)
    g1[1] = (int)(16384u << 16);    // tensor_dim0[15:0] in bits 63:48
    g1[2] = 0x00010000;             // tensor_dim0 hi=0, tensor_dim1=1
    g1[3] = (int)(16384u << 16);    // tile_dim0=16384 in bits 127:112
    g1[4] = 1;                      // tile_dim1=1, tile_dim2=0
    g1[5] = 16384;                  // tensor_dim0_stride low32
    g1[6] = (int)(16384u << 16);    // tensor_dim1_stride low16 in bits 223:208
    g1[7] = 0;
    i32x4 gz = {0, 0, 0, 0};
#if __clang_major__ >= 23
    i32x8 gz8 = {0, 0, 0, 0, 0, 0, 0, 0};
    __builtin_amdgcn_tensor_load_to_lds(g0, g1, gz, gz, gz8, 0);
#else
    __builtin_amdgcn_tensor_load_to_lds(g0, g1, gz, gz, 0);
#endif
    __builtin_amdgcn_s_wait_tensorcnt(0);
  }
#else
  for (int t = lane; t < O_ * 2 * C_; t += 32) wLds[t] = wh[t];
#endif

  for (int t = lane; t < 16 * K_; t += 32)
    nIdx[t] = knn[((size_t)b * N_ + m0) * K_ + t];
  __syncthreads();

  float bo[8];
  #pragma unroll
  for (int o = 0; o < 8; ++o) bo[o] = bias[o * 16 + lo];

  v8f rmax[8];
  #pragma unroll
  for (int o = 0; o < 8; ++o)
    #pragma unroll
    for (int r = 0; r < 8; ++r) rmax[o][r] = -3.0e38f;

  const float* xb = xt32 + (size_t)b * N_ * C_;

  for (int kbase = 0; kbase < K_; kbase += 10) {
    // Build edge features for 10 neighbor slots: 10*16*64 channel elements.
    for (int t = lane; t < 10 * 16 * C_; t += 32) {
      int c = t & (C_ - 1);
      int p = (t >> 6) & 15;
      int k = t >> 10;
      float ctr = xb[(size_t)(m0 + p) * C_ + c];
      float nb  = xb[(size_t)nIdx[p * K_ + kbase + k] * C_ + c];
      eLds[k][p][c]      = (_Float16)(nb - ctr);
      eLds[k][p][C_ + c] = (_Float16)ctr;
    }
    __syncthreads();

    #pragma unroll 1
    for (int k = 0; k < 10; ++k) {
      // A fragments from LDS for slot k (16 points x 128 features).
      Frag16 aw[4];
      const _Float16* ep = &eLds[k][lo][0];
      #pragma unroll
      for (int s = 0; s < 4; ++s) {
        aw[s].h[0] = *(const v8h*)(ep + s * 32 +      hi * 8);
        aw[s].h[1] = *(const v8h*)(ep + s * 32 + 16 + hi * 8);
      }
      #pragma unroll
      for (int o = 0; o < 8; ++o) {
        // B fragments from LDS-staged W: W[o*16+lo][table-order K].
        Frag16 bw[4];
        #pragma unroll
        for (int s = 0; s < 4; ++s) {
          const _Float16* wp = &wLds[(o * 16 + lo) * (2 * C_) + s * 32 + hi * 8];
          bw[s].h[0] = *(const v8h*)(wp);
          bw[s].h[1] = *(const v8h*)(wp + 16);
        }
        v8f c = {};
        #pragma unroll
        for (int s = 0; s < 4; ++s)
          c = __builtin_amdgcn_wmma_f32_16x16x32_f16(false, aw[s].v, false, bw[s].v,
                                                     (short)0, c, false, false);
        #pragma unroll
        for (int r = 0; r < 8; ++r) {
          float h = c[r] + bo[o];
          h = (h >= 0.f) ? h : 0.2f * h;       // leaky ReLU before max-pool
          rmax[o][r] = fmaxf(rmax[o][r], h);
        }
      }
    }
    __syncthreads();
  }

  // Store: out (B, O, N); D layout: M = r + 8*hi (point), N-col = lo (output).
  #pragma unroll
  for (int o = 0; o < 8; ++o) {
    int oo = o * 16 + lo;
    #pragma unroll
    for (int r = 0; r < 8; ++r) {
      int n = m0 + r + 8 * hi;
      out[((size_t)b * O_ + oo) * N_ + n] = rmax[o][r];
    }
  }
}

// ---------------------------------------------------------------------------
extern "C" void kernel_launch(void* const* d_in, const int* in_sizes, int n_in,
                              void* d_out, int out_size, void* d_ws, size_t ws_size,
                              hipStream_t stream) {
  const float* x    = (const float*)d_in[0];   // (B, C, N)
  const float* W    = (const float*)d_in[1];   // (O, 2C)
  const float* bias = (const float*)d_in[2];   // (O,)
  float* out = (float*)d_out;                  // (B, O, N)

  // Workspace partition (~15 MB total).
  float*    xt32 = (float*)d_ws;                             // B*N*C f32
  _Float16* xth  = (_Float16*)(xt32 + (size_t)B_ * N_ * C_); // B*N*C f16
  _Float16* wh   = xth + (size_t)B_ * N_ * C_;               // O*2C  f16
  float*    xx   = (float*)(wh + (size_t)O_ * 2 * C_);       // B*N   f32
  int*      knn  = (int*)(xx + (size_t)B_ * N_);             // B*N*K i32

  prep_x_kernel<<<(B_ * N_) / 256, 256, 0, stream>>>(x, xt32, xth, xx);
  prep_w_kernel<<<(O_ * 2 * C_) / 256, 256, 0, stream>>>(W, wh);

  knn_kernel<<<dim3(N_ / 32, B_), 32, 0, stream>>>(xth, xx, knn);
  edge_kernel<<<dim3(N_ / 16, B_), 32, 0, stream>>>(xt32, wh, bias, knn, out);
}
